// DSMIL_48026324304453
// MI455X (gfx1250) — matrix operand order repaired and checked by reading, not compiled, not run on hardware
//
#include <hip/hip_runtime.h>
#include <math.h>

// ---------------- types ----------------
typedef __bf16 bf16_t;
typedef __attribute__((ext_vector_type(16))) __bf16 v16bf;
typedef __attribute__((ext_vector_type(8)))  __bf16 v8bf;
typedef __attribute__((ext_vector_type(8)))  float   v8f;
typedef __attribute__((ext_vector_type(4)))  float   v4f;   // clang ext-vector: valid for __builtin_nontemporal_load

#define NB   4
#define NN   8192
#define NI   1024
#define ND   512
#define NC   2
#define NQ   128
#define MTOT (NB * NN)          // 32768 instances

// ---------------------------------------------------------------
// k_cvt_t: dst[n*K + k] = bf16(src[k*Nn + n])  (transpose + convert)
// ---------------------------------------------------------------
__global__ __launch_bounds__(256) void k_cvt_t(const float* __restrict__ src,
                                               bf16_t* __restrict__ dst,
                                               int K, int Nn) {
    int i = blockIdx.x * 256 + threadIdx.x;
    if (i >= K * Nn) return;
    int n = i / K, k = i % K;
    dst[i] = (bf16_t)src[k * Nn + n];
}

// ---- helpers for the encoder pipeline ----
__device__ __forceinline__ void enc_load(const float* __restrict__ xrow,
                                         const bf16_t* __restrict__ wbase,
                                         int k, int hi, v4f* a, v16bf* bm) {
    const int ka = k + hi * 16;                // hi=0: K0-15, hi=1: K16-31
    a[0] = __builtin_nontemporal_load((const v4f*)(xrow + ka));
    a[1] = __builtin_nontemporal_load((const v4f*)(xrow + ka + 4));
    a[2] = __builtin_nontemporal_load((const v4f*)(xrow + ka + 8));
    a[3] = __builtin_nontemporal_load((const v4f*)(xrow + ka + 12));
#pragma unroll
    for (int j = 0; j < 4; ++j)
        bm[j] = *(const v16bf*)(wbase + k + (long)j * 16 * NI);
}

__device__ __forceinline__ v16bf cvt_a(const v4f* a) {
    v16bf r;
    r[0]=(bf16_t)a[0].x;  r[1]=(bf16_t)a[0].y;  r[2]=(bf16_t)a[0].z;  r[3]=(bf16_t)a[0].w;
    r[4]=(bf16_t)a[1].x;  r[5]=(bf16_t)a[1].y;  r[6]=(bf16_t)a[1].z;  r[7]=(bf16_t)a[1].w;
    r[8]=(bf16_t)a[2].x;  r[9]=(bf16_t)a[2].y;  r[10]=(bf16_t)a[2].z; r[11]=(bf16_t)a[2].w;
    r[12]=(bf16_t)a[3].x; r[13]=(bf16_t)a[3].y; r[14]=(bf16_t)a[3].z; r[15]=(bf16_t)a[3].w;
    return r;
}

// ---------------------------------------------------------------
// k_enc: h[m, n] = bf16(relu( x[m,:]@W_enc[:,n] + b_enc[n] ))
// block = 8 waves; block -> 16-row M tile; wave w -> 64 columns (4 wmma)
// Ping-pong double-buffered pipeline (K unrolled x2): no register rotation,
// no WMMA->VALU WAR hazards. x streamed non-temporal; h/weights L2-resident.
// ---------------------------------------------------------------
__global__ __launch_bounds__(256) void k_enc(const float* __restrict__ x,
                                             const bf16_t* __restrict__ Wt,   // [ND][NI] bf16 (n-major)
                                             const float* __restrict__ b_enc,
                                             bf16_t* __restrict__ h) {
    const int tid  = threadIdx.x;
    const int wave = tid >> 5;
    const int lane = tid & 31;
    const int l16  = lane & 15;
    const int hi   = lane >> 4;
    const long mbase = (long)blockIdx.x * 16;
    const long row   = mbase + l16;

    const float*  xrow  = x + row * NI;
    const bf16_t* wbase = Wt + (long)(wave * 64 + l16) * NI + hi * 16;

    v8f acc[4];
#pragma unroll
    for (int j = 0; j < 4; ++j) acc[j] = (v8f){0,0,0,0,0,0,0,0};

    v4f  pa[2][4];
    v16bf pb[2][4];
    enc_load(xrow, wbase, 0, hi, pa[0], pb[0]);

    for (int k0 = 0; k0 < NI; k0 += 64) {
        // half 1: prefetch k0+32 into buf1, compute from buf0
        enc_load(xrow, wbase, k0 + 32, hi, pa[1], pb[1]);
        {
            const v16bf a = cvt_a(pa[0]);
#pragma unroll
            for (int j = 0; j < 4; ++j)
                acc[j] = __builtin_amdgcn_wmma_f32_16x16x32_bf16(
                    false, a, false, pb[0][j], (short)0, acc[j], false, false);
        }
        // half 2: prefetch k0+64 (wrap on last iter) into buf0, compute from buf1
        const int k2 = (k0 + 64 < NI) ? (k0 + 64) : 0;
        enc_load(xrow, wbase, k2, hi, pa[0], pb[0]);
        {
            const v16bf a = cvt_a(pa[1]);
#pragma unroll
            for (int j = 0; j < 4; ++j)
                acc[j] = __builtin_amdgcn_wmma_f32_16x16x32_bf16(
                    false, a, false, pb[1][j], (short)0, acc[j], false, false);
        }
    }

    // ---- epilogue: bias + ReLU, store bf16 (C layout: lanes<16 rows0-7) ----
#pragma unroll
    for (int j = 0; j < 4; ++j) {
        const int n = wave * 64 + j * 16 + l16;
        const float bias = b_enc[n];
#pragma unroll
        for (int r = 0; r < 8; ++r) {
            const int rr = hi * 8 + r;
            float v = acc[j][r] + bias;
            v = v > 0.f ? v : 0.f;
            h[(mbase + rr) * ND + n] = (bf16_t)v;
        }
    }
}

// ---------------------------------------------------------------
// k_cls: instance scores + per-block argmax. wave-per-instance.
// grid = MTOT/8 = 4096 blocks (1024 per bag)
// ---------------------------------------------------------------
__global__ __launch_bounds__(256) void k_cls(const bf16_t* __restrict__ h,
                                             const float* __restrict__ W_i,
                                             const float* __restrict__ b_i,
                                             float* __restrict__ pm_val,
                                             int*   __restrict__ pm_idx) {
    const int tid  = threadIdx.x;
    const int wave = tid >> 5;
    const int lane = tid & 31;
    const long g = (long)blockIdx.x * 8 + wave;
    const int  n = (int)(g & (NN - 1));
    float s0 = 0.f, s1 = 0.f;
    for (int d = lane; d < ND; d += 32) {
        const float hv = (float)h[g * ND + d];
        s0 += hv * W_i[d * 2 + 0];
        s1 += hv * W_i[d * 2 + 1];
    }
#pragma unroll
    for (int m = 16; m >= 1; m >>= 1) {
        s0 += __shfl_xor(s0, m, 32);
        s1 += __shfl_xor(s1, m, 32);
    }
    __shared__ float sv[2][8];
    __shared__ int   si[8];
    if (lane == 0) { sv[0][wave] = s0 + b_i[0]; sv[1][wave] = s1 + b_i[1]; si[wave] = n; }
    __syncthreads();
    if (tid < 2) {
        const int c = tid;
        float best = sv[c][0]; int bidx = si[0];
        for (int w = 1; w < 8; ++w) {
            if (sv[c][w] > best) { best = sv[c][w]; bidx = si[w]; }   // first-index tie-break
        }
        const int b   = blockIdx.x >> 10;
        const int blk = blockIdx.x & 1023;
        pm_val[(b * 2 + c) * 1024 + blk] = best;
        pm_idx[(b * 2 + c) * 1024 + blk] = bidx;
    }
}

// ---------------------------------------------------------------
// k_qmax: final argmax + q_fn on the 8 critical rows (single block)
// ---------------------------------------------------------------
__global__ __launch_bounds__(256) void k_qmax(const bf16_t* __restrict__ h,
                                              const float* __restrict__ pm_val,
                                              const int*   __restrict__ pm_idx,
                                              const float* __restrict__ W_q1,
                                              const float* __restrict__ b_q1,
                                              const float* __restrict__ W_q2,
                                              const float* __restrict__ b_q2,
                                              float* __restrict__ qmax) {
    __shared__ float red[256];
    __shared__ int   redi[256];
    __shared__ int   best_idx[8];
    __shared__ float z[8][NQ];
    const int tid = threadIdx.x;
    for (int p = 0; p < 8; ++p) {
        float best = -INFINITY; int bidx = 0x7fffffff;
        for (int i = tid; i < 1024; i += 256) {
            const float v = pm_val[p * 1024 + i];
            const int  ix = pm_idx[p * 1024 + i];
            if (v > best || (v == best && ix < bidx)) { best = v; bidx = ix; }
        }
        red[tid] = best; redi[tid] = bidx;
        __syncthreads();
        for (int s = 128; s > 0; s >>= 1) {
            if (tid < s) {
                if (red[tid + s] > red[tid] ||
                    (red[tid + s] == red[tid] && redi[tid + s] < redi[tid])) {
                    red[tid] = red[tid + s]; redi[tid] = redi[tid + s];
                }
            }
            __syncthreads();
        }
        if (tid == 0) best_idx[p] = redi[0];
        __syncthreads();
    }
    // q_fn(m_feats): z = relu(h_row @ W_q1 + b_q1)
    for (int p = 0; p < 8; ++p) {
        const int b = p >> 1;
        const long g = (long)b * NN + best_idx[p];
        if (tid < NQ) {
            float s = 0.f;
            for (int d = 0; d < ND; ++d) s += (float)h[g * ND + d] * W_q1[d * NQ + tid];
            const float zz = s + b_q1[tid];
            z[p][tid] = zz > 0.f ? zz : 0.f;
        }
    }
    __syncthreads();
    // qmax = tanh(z @ W_q2 + b_q2)
    for (int p = 0; p < 8; ++p) {
        if (tid < NQ) {
            float s = 0.f;
            for (int q = 0; q < NQ; ++q) s += z[p][q] * W_q2[q * NQ + tid];
            qmax[p * NQ + tid] = tanhf(s + b_q2[tid]);
        }
    }
}

// ---------------------------------------------------------------
// k_q: Q = tanh(relu(h@W_q1 + b1) @ W_q2 + b2), two chained WMMA GEMMs
// block -> 16-row tile, wave w -> columns [16w,16w+16)
// GEMM1 ping-pong double-buffered like k_enc.
// ---------------------------------------------------------------
#define ZPITCH 136   // 128 + 8 bf16 pad -> conflict-free 16B LDS reads
__device__ __forceinline__ void q_load(const bf16_t* __restrict__ hrow,
                                       const bf16_t* __restrict__ w1base,
                                       int k, int hi, v8bf& ax, v8bf& ay, v16bf& bw) {
    const int ka = k + hi * 16;
    ax = *(const v8bf*)(hrow + ka);
    ay = *(const v8bf*)(hrow + ka + 8);
    bw = *(const v16bf*)(w1base + k);
}

__global__ __launch_bounds__(256) void k_q(const bf16_t* __restrict__ h,
                                           const bf16_t* __restrict__ Wq1t,  // [NQ][ND]
                                           const float* __restrict__ b_q1,
                                           const bf16_t* __restrict__ Wq2t,  // [NQ][NQ]
                                           const float* __restrict__ b_q2,
                                           bf16_t* __restrict__ Q) {
    __shared__ alignas(16) bf16_t zs[16 * ZPITCH];
    const int tid  = threadIdx.x;
    const int wave = tid >> 5;
    const int lane = tid & 31;
    const int l16  = lane & 15;
    const int hi   = lane >> 4;
    const long mbase = (long)blockIdx.x * 16;
    const long row   = mbase + l16;

    const bf16_t* hrow   = h + row * ND;
    const bf16_t* w1base = Wq1t + (long)(wave * 16 + l16) * ND + hi * 16;

    // ---- GEMM1: z = relu(h @ W_q1 + b_q1), K=512, ping-pong pipelined ----
    v8f acc = (v8f){0,0,0,0,0,0,0,0};
    v8bf ax[2], ay[2]; v16bf bw[2];
    q_load(hrow, w1base, 0, hi, ax[0], ay[0], bw[0]);
    for (int k0 = 0; k0 < ND; k0 += 64) {
        q_load(hrow, w1base, k0 + 32, hi, ax[1], ay[1], bw[1]);
        {
            v16bf a;
#pragma unroll
            for (int i = 0; i < 8; ++i) { a[i] = ax[0][i]; a[8 + i] = ay[0][i]; }
            acc = __builtin_amdgcn_wmma_f32_16x16x32_bf16(
                false, a, false, bw[0], (short)0, acc, false, false);
        }
        const int k2 = (k0 + 64 < ND) ? (k0 + 64) : 0;
        q_load(hrow, w1base, k2, hi, ax[0], ay[0], bw[0]);
        {
            v16bf a;
#pragma unroll
            for (int i = 0; i < 8; ++i) { a[i] = ax[1][i]; a[8 + i] = ay[1][i]; }
            acc = __builtin_amdgcn_wmma_f32_16x16x32_bf16(
                false, a, false, bw[1], (short)0, acc, false, false);
        }
    }
    {
        const int n = wave * 16 + l16;
        const float bias = b_q1[n];
#pragma unroll
        for (int r = 0; r < 8; ++r) {
            const int rr = hi * 8 + r;
            float v = acc[r] + bias;
            v = v > 0.f ? v : 0.f;
            zs[rr * ZPITCH + n] = (bf16_t)v;
        }
    }
    __syncthreads();

    // ---- GEMM2: out = tanh(z @ W_q2 + b_q2), K=128, A from LDS ----
    v8f acc2 = (v8f){0,0,0,0,0,0,0,0};
#pragma unroll
    for (int k0 = 0; k0 < NQ; k0 += 32) {
        const int ka = k0 + hi * 16;
        const v8bf lx = *(const v8bf*)(zs + l16 * ZPITCH + ka);
        const v8bf ly = *(const v8bf*)(zs + l16 * ZPITCH + ka + 8);
        v16bf a;
#pragma unroll
        for (int i = 0; i < 8; ++i) { a[i] = lx[i]; a[8 + i] = ly[i]; }
        const v16bf bm = *(const v16bf*)(Wq2t + (long)(wave * 16 + l16) * NQ + k0 + hi * 16);
        acc2 = __builtin_amdgcn_wmma_f32_16x16x32_bf16(
            false, a, false, bm, (short)0, acc2, false, false);
    }
    {
        const int n = wave * 16 + l16;
        const float bias = b_q2[n];
#pragma unroll
        for (int r = 0; r < 8; ++r) {
            const int rr = hi * 8 + r;
            Q[(mbase + rr) * NQ + n] = (bf16_t)tanhf(acc2[r] + bias);
        }
    }
}

// ---------------------------------------------------------------
// k_logits: l[g,c] = Q[g,:].qmax[b,c,:] / sqrt(128); per-block max
// wave-per-instance; grid = 4096 blocks (1024 per bag)
// ---------------------------------------------------------------
__global__ __launch_bounds__(256) void k_logits(const bf16_t* __restrict__ Q,
                                                const float* __restrict__ qmax,
                                                float* __restrict__ lbuf,
                                                float* __restrict__ pmax) {
    const int tid  = threadIdx.x;
    const int wave = tid >> 5;
    const int lane = tid & 31;
    const long g = (long)blockIdx.x * 8 + wave;
    const int  b = (int)(g >> 13);
    const float* qm0 = qmax + (b * 2 + 0) * NQ;
    const float* qm1 = qmax + (b * 2 + 1) * NQ;
    float s0 = 0.f, s1 = 0.f;
#pragma unroll
    for (int k = 0; k < 4; ++k) {
        const int d  = lane + 32 * k;
        const float qv = (float)Q[g * NQ + d];
        s0 += qv * qm0[d];
        s1 += qv * qm1[d];
    }
#pragma unroll
    for (int m = 16; m >= 1; m >>= 1) {
        s0 += __shfl_xor(s0, m, 32);
        s1 += __shfl_xor(s1, m, 32);
    }
    __shared__ float sm[2][8];
    const float scale = 0.08838834764831845f;   // 1/sqrt(128)
    if (lane == 0) {
        s0 *= scale; s1 *= scale;
        lbuf[g * 2 + 0] = s0; lbuf[g * 2 + 1] = s1;
        sm[0][wave] = s0; sm[1][wave] = s1;
    }
    __syncthreads();
    if (tid < 2) {
        float m2 = sm[tid][0];
        for (int w = 1; w < 8; ++w) m2 = fmaxf(m2, sm[tid][w]);
        pmax[(b * 2 + tid) * 1024 + (blockIdx.x & 1023)] = m2;
    }
}

__global__ __launch_bounds__(256) void k_redmax(const float* __restrict__ pmax,
                                                float* __restrict__ Mv) {
    __shared__ float red[256];
    const int tid = threadIdx.x;
    for (int p = 0; p < 8; ++p) {
        float m = -INFINITY;
        for (int i = tid; i < 1024; i += 256) m = fmaxf(m, pmax[p * 1024 + i]);
        red[tid] = m;
        __syncthreads();
        for (int s = 128; s > 0; s >>= 1) {
            if (tid < s) red[tid] = fmaxf(red[tid], red[tid + s]);
            __syncthreads();
        }
        if (tid == 0) Mv[p] = red[0];
        __syncthreads();
    }
}

// exp in place + per-block sums; grid = MTOT/256 = 128 (32 per bag)
__global__ __launch_bounds__(256) void k_expsum(float* __restrict__ lbuf,
                                                const float* __restrict__ Mv,
                                                float* __restrict__ psum) {
    const int tid = threadIdx.x;
    const long g = (long)blockIdx.x * 256 + tid;
    const int  b = (int)(g >> 13);
    const float e0 = expf(lbuf[g * 2 + 0] - Mv[b * 2 + 0]);
    const float e1 = expf(lbuf[g * 2 + 1] - Mv[b * 2 + 1]);
    lbuf[g * 2 + 0] = e0; lbuf[g * 2 + 1] = e1;
    __shared__ float r0[256], r1[256];
    r0[tid] = e0; r1[tid] = e1;
    __syncthreads();
    for (int s = 128; s > 0; s >>= 1) {
        if (tid < s) { r0[tid] += r0[tid + s]; r1[tid] += r1[tid + s]; }
        __syncthreads();
    }
    if (tid == 0) {
        const int blk = blockIdx.x & 31;
        psum[(b * 2 + 0) * 32 + blk] = r0[0];
        psum[(b * 2 + 1) * 32 + blk] = r1[0];
    }
}

__global__ __launch_bounds__(256) void k_redsum(const float* __restrict__ psum,
                                                float* __restrict__ Sv) {
    const int tid = threadIdx.x;
    const int pair = tid >> 5, i = tid & 31;
    float v = (pair < 8) ? psum[pair * 32 + i] : 0.f;
#pragma unroll
    for (int m = 16; m >= 1; m >>= 1) v += __shfl_xor(v, m, 32);
    if (i == 0 && pair < 8) Sv[pair] = v;
}

__global__ __launch_bounds__(256) void k_zero(float* __restrict__ out, int n) {
    const int i = blockIdx.x * 256 + threadIdx.x;
    if (i < n) out[i] = 0.f;
}

// B_out[b,c,d] = sum_n (e[g,c]/S) * h[g,d]; grid = NB*64, 128 instances/block
__global__ __launch_bounds__(256) void k_out(const bf16_t* __restrict__ h,
                                             const float* __restrict__ ebuf,
                                             const float* __restrict__ Sv,
                                             float* __restrict__ out) {
    const int b  = blockIdx.x >> 6;
    const int ns = blockIdx.x & 63;
    const int tid = threadIdx.x;
    const float inv0 = 1.f / Sv[b * 2 + 0];
    const float inv1 = 1.f / Sv[b * 2 + 1];
    const int d0 = tid * 2;
    float a00 = 0.f, a01 = 0.f, a10 = 0.f, a11 = 0.f;
    for (int i = 0; i < 128; ++i) {
        const long g = (long)b * NN + ns * 128 + i;
        const float w0 = ebuf[g * 2 + 0] * inv0;
        const float w1 = ebuf[g * 2 + 1] * inv1;
        const float h0 = (float)h[g * ND + d0];
        const float h1 = (float)h[g * ND + d0 + 1];
        a00 += w0 * h0; a01 += w0 * h1;
        a10 += w1 * h0; a11 += w1 * h1;
    }
    atomicAdd(&out[(b * 2 + 0) * ND + d0],     a00);
    atomicAdd(&out[(b * 2 + 0) * ND + d0 + 1], a01);
    atomicAdd(&out[(b * 2 + 1) * ND + d0],     a10);
    atomicAdd(&out[(b * 2 + 1) * ND + d0 + 1], a11);
}

// ---------------------------------------------------------------
extern "C" void kernel_launch(void* const* d_in, const int* in_sizes, int n_in,
                              void* d_out, int out_size, void* d_ws, size_t ws_size,
                              hipStream_t stream) {
    const float* x     = (const float*)d_in[0];
    const float* W_enc = (const float*)d_in[1];
    const float* b_enc = (const float*)d_in[2];
    const float* W_i   = (const float*)d_in[3];
    const float* b_i   = (const float*)d_in[4];
    const float* W_q1  = (const float*)d_in[5];
    const float* b_q1  = (const float*)d_in[6];
    const float* W_q2  = (const float*)d_in[7];
    const float* b_q2  = (const float*)d_in[8];
    float* out = (float*)d_out;

    char* ws = (char*)d_ws;
    size_t off = 0;
    auto carve = [&](size_t bytes) -> char* {
        char* p = ws + off;
        off = (off + bytes + 255) & ~(size_t)255;
        return p;
    };
    bf16_t* Wenc_t = (bf16_t*)carve((size_t)ND * NI * 2);   // [512][1024]
    bf16_t* Wq1_t  = (bf16_t*)carve((size_t)NQ * ND * 2);   // [128][512]
    bf16_t* Wq2_t  = (bf16_t*)carve((size_t)NQ * NQ * 2);   // [128][128]
    bf16_t* hbuf   = (bf16_t*)carve((size_t)MTOT * ND * 2); // 33.5 MB
    bf16_t* Qbuf   = (bf16_t*)carve((size_t)MTOT * NQ * 2); // 8.4 MB
    float*  lbuf   = (float*)carve((size_t)MTOT * 2 * 4);
    float*  pm_val = (float*)carve(8 * 1024 * 4);
    int*    pm_idx = (int*)carve(8 * 1024 * 4);
    float*  qmax   = (float*)carve(8 * NQ * 4);
    float*  pmax   = (float*)carve(8 * 1024 * 4);
    float*  Mv     = (float*)carve(8 * 4);
    float*  psum   = (float*)carve(8 * 32 * 4);
    float*  Sv     = (float*)carve(8 * 4);

    // 0) weight transpose+convert to bf16
    k_cvt_t<<<(NI * ND + 255) / 256, 256, 0, stream>>>(W_enc, Wenc_t, NI, ND);
    k_cvt_t<<<(ND * NQ + 255) / 256, 256, 0, stream>>>(W_q1, Wq1_t, ND, NQ);
    k_cvt_t<<<(NQ * NQ + 255) / 256, 256, 0, stream>>>(W_q2, Wq2_t, NQ, NQ);
    // zero output (atomics target)
    k_zero<<<(NB * NC * ND + 255) / 256, 256, 0, stream>>>(out, NB * NC * ND);

    // 1) encoder GEMM (WMMA bf16, fp32 accumulate, ping-pong pipeline, NT x-stream)
    k_enc<<<MTOT / 16, 256, 0, stream>>>(x, Wenc_t, b_enc, hbuf);
    // 2) instance scores + argmax + q_fn on critical rows
    k_cls<<<MTOT / 8, 256, 0, stream>>>(hbuf, W_i, b_i, pm_val, pm_idx);
    k_qmax<<<1, 256, 0, stream>>>(hbuf, pm_val, pm_idx, W_q1, b_q1, W_q2, b_q2, qmax);
    // 3) Q = q_fn(h) (two chained WMMA GEMMs)
    k_q<<<MTOT / 16, 256, 0, stream>>>(hbuf, Wq1_t, b_q1, Wq2_t, b_q2, Qbuf);
    // 4) attention softmax over instances
    k_logits<<<MTOT / 8, 256, 0, stream>>>(Qbuf, qmax, lbuf, pmax);
    k_redmax<<<1, 256, 0, stream>>>(pmax, Mv);
    k_expsum<<<MTOT / 256, 256, 0, stream>>>(lbuf, Mv, psum);
    k_redsum<<<1, 256, 0, stream>>>(psum, Sv);
    // 5) B_out = A^T @ h
    k_out<<<NB * 64, 256, 0, stream>>>(hbuf, lbuf, Sv, out);
    (void)in_sizes; (void)n_in; (void)out_size; (void)ws_size;
}
